// PatchCoreModel_2534030704994
// MI455X (gfx1250) — matrix operand order repaired and checked by reading, not compiled, not run on hardware
//
#include <hip/hip_runtime.h>
#include <hip/hip_bf16.h>

typedef __attribute__((ext_vector_type(16))) __bf16 v16bf;
typedef __attribute__((ext_vector_type(8)))  float  v8f;

#define N_ROWS   6272      // 8*28*28 feature rows
#define DIM      128
#define M_ROWS   30000
#define SEG      16
#define SEG_LEN  1875      // 30000 / 16
#define CHUNK    64        // memory-bank rows staged in LDS per iteration
#define LDS_STR  136       // padded row stride in bf16 elems: 272B -> 4-bank skew per row
#define BATCH    8
#define PH       28
#define IMG      224

union PackU { __bf16 h[8];  uint4 u;    };   // 8 bf16 -> one 16B store
union LoadU { uint4 u[2];   v16bf v;    };   // two 16B loads -> one WMMA operand

// ---------------------------------------------------------------- f2 = ||f||^2
__global__ __launch_bounds__(256) void rowsumsq_kernel(const float* __restrict__ F,
                                                       float* __restrict__ f2) {
  int r = blockIdx.x * blockDim.x + threadIdx.x;
  if (r >= N_ROWS) return;
  const float4* p = (const float4*)(F + (size_t)r * DIM);
  float s = 0.f;
#pragma unroll
  for (int i = 0; i < DIM / 4; ++i) {
    float4 v = p[i];
    s += v.x * v.x + v.y * v.y + v.z * v.z + v.w * v.w;
  }
  f2[r] = s;
}

// ------------------------------------------------- WMMA distance + row-min kernel
// D = F x MB^T via error-compensated bf16 split (hi*hi + hi*lo + lo*hi), f32 accum.
// Each block: 128 feature rows (8 waves x 16 rows), one M-segment of 1875 rows.
__global__ __launch_bounds__(256) void nnmin_kernel(const float* __restrict__ F,
                                                    const float* __restrict__ MB,
                                                    float* __restrict__ nnpart) {
  __shared__ __bf16 ldsBhi[CHUNK * LDS_STR];
  __shared__ __bf16 ldsBlo[CHUNK * LDS_STR];
  __shared__ float  ldsPart[256];
  __shared__ float  ldsM2[CHUNK];

  const int tid   = threadIdx.x;
  const int lane  = tid & 31;
  const int wave  = tid >> 5;
  const int halfH = lane >> 4;        // lane half: selects K sub-ranges / C rows 0-7 vs 8-15
  const int l15   = lane & 15;

  const int blockRow = blockIdx.x * 128;
  const int segStart = blockIdx.y * SEG_LEN;
  const int segEnd   = segStart + SEG_LEN;

  // ---- load + split A tile for this wave (ISA 16-bit A 16x32 layout, 4 K-blocks)
  // lane half h, element e<8 -> K = 32*kb + 8h + e ; e>=8 -> K = 32*kb + 16 + 8h + (e-8)
  const int rowA = blockRow + wave * 16 + l15;
  v16bf Ahi[4], Alo[4];
  {
    const float* frow = F + (size_t)rowA * DIM;
#pragma unroll
    for (int kb = 0; kb < 4; ++kb) {
      const float4* p0 = (const float4*)(frow + 32 * kb + 8 * halfH);
      const float4* p1 = (const float4*)(frow + 32 * kb + 16 + 8 * halfH);
      float4 q0 = p0[0], q1 = p0[1], q2 = p1[0], q3 = p1[1];
      float rr[16] = {q0.x, q0.y, q0.z, q0.w, q1.x, q1.y, q1.z, q1.w,
                      q2.x, q2.y, q2.z, q2.w, q3.x, q3.y, q3.z, q3.w};
#pragma unroll
      for (int e = 0; e < 16; ++e) {
        __bf16 h = (__bf16)rr[e];
        Ahi[kb][e] = h;
        Alo[kb][e] = (__bf16)(rr[e] - (float)h);
      }
    }
  }

  float minv[8];
#pragma unroll
  for (int v = 0; v < 8; ++v) minv[v] = __builtin_inff();

  for (int base = segStart; base < segEnd; base += CHUNK) {
    __syncthreads();   // previous tile compute done before LDS overwrite

    // ---- stage 64 memory rows: fp32 -> bf16 hi/lo into LDS, fold sumsq
    {
      const int r = tid >> 2;
      const int q = tid & 3;
      const int gRow = base + r;
      const int off = r * LDS_STR + q * 32;      // bf16 elements
      float psum = 0.f;
      if (gRow < M_ROWS) {
        const float4* src = (const float4*)(MB + (size_t)gRow * DIM) + q * 8;
#pragma unroll
        for (int g = 0; g < 4; ++g) {
          float4 a = src[2 * g], b = src[2 * g + 1];
          float f[8] = {a.x, a.y, a.z, a.w, b.x, b.y, b.z, b.w};
          PackU ph, pl;
#pragma unroll
          for (int e = 0; e < 8; ++e) {
            __bf16 h = (__bf16)f[e];
            ph.h[e] = h;
            pl.h[e] = (__bf16)(f[e] - (float)h);
            psum += f[e] * f[e];
          }
          *(uint4*)&ldsBhi[off + g * 8] = ph.u;
          *(uint4*)&ldsBlo[off + g * 8] = pl.u;
        }
      } else {
        uint4 z = make_uint4(0u, 0u, 0u, 0u);
#pragma unroll
        for (int g = 0; g < 4; ++g) {
          *(uint4*)&ldsBhi[off + g * 8] = z;
          *(uint4*)&ldsBlo[off + g * 8] = z;
        }
      }
      ldsPart[tid] = psum;
    }
    __syncthreads();
    if (tid < CHUNK)
      ldsM2[tid] = ldsPart[tid * 4] + ldsPart[tid * 4 + 1] +
                   ldsPart[tid * 4 + 2] + ldsPart[tid * 4 + 3];

    // speculative prefetch of next chunk (global_prefetch_b8)
    if (base + CHUNK < segEnd) {
      const char* nxt = (const char*)(MB + (size_t)(base + CHUNK) * DIM) + tid * 128;
      __builtin_prefetch(nxt, 0, 1);
    }
    __syncthreads();

    // ---- 4 column tiles of 16; K=128 -> 4 K-blocks x 3 split terms = 12 WMMA per tile
    // B layout (16-bit 32x16): lane = column, lane half h covers K in [8h, 8h+8) and [16+8h, 16+8h+8)
#pragma unroll
    for (int t = 0; t < 4; ++t) {
      const int colLocal = t * 16 + l15;
      const int gCol = base + colLocal;
      const __bf16* bh = &ldsBhi[colLocal * LDS_STR + 8 * halfH];
      const __bf16* bl = &ldsBlo[colLocal * LDS_STR + 8 * halfH];
      v8f aHH = {}, aHL = {}, aLH = {};
#pragma unroll
      for (int kb = 0; kb < 4; ++kb) {
        LoadU bhv, blv;
        bhv.u[0] = *(const uint4*)(bh + 32 * kb);
        bhv.u[1] = *(const uint4*)(bh + 32 * kb + 16);
        blv.u[0] = *(const uint4*)(bl + 32 * kb);
        blv.u[1] = *(const uint4*)(bl + 32 * kb + 16);
        aHH = __builtin_amdgcn_wmma_f32_16x16x32_bf16(false, Ahi[kb], false, bhv.v,
                                                      (short)0, aHH, false, false);
        aHL = __builtin_amdgcn_wmma_f32_16x16x32_bf16(false, Ahi[kb], false, blv.v,
                                                      (short)0, aHL, false, false);
        aLH = __builtin_amdgcn_wmma_f32_16x16x32_bf16(false, Alo[kb], false, bhv.v,
                                                      (short)0, aLH, false, false);
      }
      const float m2v = (gCol < M_ROWS) ? ldsM2[colLocal] : __builtin_inff();
#pragma unroll
      for (int v = 0; v < 8; ++v) {
        float dot = aHH[v] + (aHL[v] + aLH[v]);
        float d = fmaf(-2.f, dot, m2v);            // m^2 - 2*f.m
        minv[v] = fminf(minv[v], d);
      }
    }
  }

  // ---- min across the 16 lanes of each half (C layout: lane = column)
#pragma unroll
  for (int m = 8; m >= 1; m >>= 1) {
#pragma unroll
    for (int v = 0; v < 8; ++v)
      minv[v] = fminf(minv[v], __shfl_xor(minv[v], m, 32));
  }

  if (l15 == 0) {  // lanes 0 (rows 0-7) and 16 (rows 8-15) of each wave
    float* out = nnpart + (size_t)blockIdx.y * N_ROWS + blockRow + wave * 16 + halfH * 8;
#pragma unroll
    for (int v = 0; v < 8; ++v) out[v] = minv[v];
  }
}

// ---------------------------------- combine segment minima + f2, per-image max
__global__ __launch_bounds__(256) void combine_max_kernel(const float* __restrict__ f2,
                                                          const float* __restrict__ nnpart,
                                                          float* __restrict__ nn,
                                                          float* __restrict__ img_scores) {
  __shared__ float red[256];
  const int b = blockIdx.x;
  float lmax = -__builtin_inff();
  for (int p = threadIdx.x; p < PH * PH; p += blockDim.x) {
    int r = b * PH * PH + p;
    float mn = __builtin_inff();
#pragma unroll
    for (int s = 0; s < SEG; ++s) mn = fminf(mn, nnpart[(size_t)s * N_ROWS + r]);
    float val = mn + f2[r];
    nn[r] = val;
    lmax = fmaxf(lmax, val);
  }
  red[threadIdx.x] = lmax;
  __syncthreads();
  for (int s = 128; s > 0; s >>= 1) {
    if (threadIdx.x < s) red[threadIdx.x] = fmaxf(red[threadIdx.x], red[threadIdx.x + s]);
    __syncthreads();
  }
  if (threadIdx.x == 0) img_scores[b] = red[0];
}

// --------------------------- half-pixel bilinear upsample 28x28 -> 224x224
__global__ __launch_bounds__(256) void upsample_kernel(const float* __restrict__ nn,
                                                       float* __restrict__ up) {
  int idx = blockIdx.x * blockDim.x + threadIdx.x;
  if (idx >= BATCH * IMG * IMG) return;
  int b   = idx / (IMG * IMG);
  int rem = idx % (IMG * IMG);
  int oy  = rem / IMG;
  int ox  = rem % IMG;
  const float scale = (float)PH / (float)IMG;      // 0.125
  float sy = (oy + 0.5f) * scale - 0.5f;
  float sx = (ox + 0.5f) * scale - 0.5f;
  int y0 = (int)floorf(sy), x0 = (int)floorf(sx);
  float wy = sy - (float)y0, wx = sx - (float)x0;
  int y0c = min(max(y0, 0), PH - 1), y1c = min(max(y0 + 1, 0), PH - 1);
  int x0c = min(max(x0, 0), PH - 1), x1c = min(max(x0 + 1, 0), PH - 1);
  const float* p = nn + b * PH * PH;
  float v00 = p[y0c * PH + x0c], v01 = p[y0c * PH + x1c];
  float v10 = p[y1c * PH + x0c], v11 = p[y1c * PH + x1c];
  float v0 = v00 + (v01 - v00) * wx;
  float v1 = v10 + (v11 - v10) * wx;
  up[idx] = v0 + (v1 - v0) * wy;
}

extern "C" void kernel_launch(void* const* d_in, const int* in_sizes, int n_in,
                              void* d_out, int out_size, void* d_ws, size_t ws_size,
                              hipStream_t stream) {
  const float* F  = (const float*)d_in[0];   // features  [6272,128] f32
  const float* MB = (const float*)d_in[1];   // memory    [30000,128] f32
  float* out = (float*)d_out;                // [8] image scores ++ [8,224,224] map

  float* f2     = (float*)d_ws;                       // N floats
  float* nnpart = f2 + N_ROWS;                        // SEG*N floats
  float* nn     = nnpart + (size_t)SEG * N_ROWS;      // N floats

  rowsumsq_kernel<<<(N_ROWS + 255) / 256, 256, 0, stream>>>(F, f2);

  dim3 grid(N_ROWS / 128, SEG);                       // 49 x 16 = 784 blocks
  nnmin_kernel<<<grid, 256, 0, stream>>>(F, MB, nnpart);

  combine_max_kernel<<<BATCH, 256, 0, stream>>>(f2, nnpart, nn, out);

  upsample_kernel<<<(BATCH * IMG * IMG + 255) / 256, 256, 0, stream>>>(nn, out + BATCH);
}
